// SingleMambaBlock_6854767804606
// MI455X (gfx1250) — matrix operand order repaired and checked by reading, not compiled, not run on hardware
//
#include <hip/hip_runtime.h>
#include <hip/hip_bf16.h>
#include <cstdint>
#include <cstddef>

// ---------------------------------------------------------------------------
// Problem constants (match reference)
// ---------------------------------------------------------------------------
#define BB      4
#define LL      2048
#define DD      1024
#define EE      2048      // D_INNER
#define NSTATE  16
#define DCONV   4
#define DTRANK  64
#define MM      (BB * LL) // 8192 rows

typedef __bf16 bf16_t;
typedef __attribute__((ext_vector_type(16))) bf16_t bf16x16;
typedef __attribute__((ext_vector_type(8)))  bf16_t bf16x8;
typedef __attribute__((ext_vector_type(8)))  float  f32x8;
typedef __attribute__((ext_vector_type(4)))  int    v4i;

typedef __attribute__((address_space(1))) v4i gv4i;   // global int32x4
typedef __attribute__((address_space(3))) v4i lv4i;   // LDS int32x4

// 16-byte async global -> LDS copy (ASYNCcnt-tracked)
__device__ inline void async_cp16(const bf16_t* g, bf16_t* l) {
    __builtin_amdgcn_global_load_async_to_lds_b128(
        (gv4i*)g, (lv4i*)l, /*imm offset*/0, /*cpol*/0);
}

// ---------------------------------------------------------------------------
// fp32 -> bf16 conversion (weights)
// ---------------------------------------------------------------------------
__global__ void cvt_bf16_kernel(const float* __restrict__ in,
                                bf16_t* __restrict__ out, int n) {
    int i = blockIdx.x * blockDim.x + threadIdx.x;
    if (i < n) out[i] = (bf16_t)in[i];
}

// ---------------------------------------------------------------------------
// residual add + LayerNorm(D=1024) -> residual (f32, to d_out tail) + xn bf16
// ---------------------------------------------------------------------------
__global__ __launch_bounds__(256) void add_ln_kernel(
    const float* __restrict__ x, const float* __restrict__ res_in,
    const float* __restrict__ w, const float* __restrict__ b,
    float* __restrict__ res_out, bf16_t* __restrict__ xn) {
    __shared__ float red[256];
    __shared__ float red2[256];
    const size_t base = (size_t)blockIdx.x * DD;
    float v[4];
    float s = 0.f, s2 = 0.f;
#pragma unroll
    for (int j = 0; j < 4; ++j) {
        int i = threadIdx.x + j * 256;
        float t = x[base + i] + res_in[base + i];
        res_out[base + i] = t;
        v[j] = t; s += t; s2 += t * t;
    }
    red[threadIdx.x] = s; red2[threadIdx.x] = s2;
    __syncthreads();
    for (int off = 128; off > 0; off >>= 1) {
        if (threadIdx.x < off) {
            red[threadIdx.x]  += red[threadIdx.x + off];
            red2[threadIdx.x] += red2[threadIdx.x + off];
        }
        __syncthreads();
    }
    float mean = red[0] * (1.0f / DD);
    float var  = red2[0] * (1.0f / DD) - mean * mean;
    float inv  = rsqrtf(var + 1e-5f);
#pragma unroll
    for (int j = 0; j < 4; ++j) {
        int i = threadIdx.x + j * 256;
        xn[base + i] = (bf16_t)((v[j] - mean) * inv * w[i] + b[i]);
    }
}

// ---------------------------------------------------------------------------
// bf16 WMMA GEMM with double-buffered async global->LDS staging.
//   C[M,N] (f32) = A[M,K] (bf16) * W[N,K]^T (bf16)
// WG = 256 thr (8 waves). C tile / WG = 128(M) x 64(N), BK = 32.
// LDS: A tile 128x32 (8KB), B tile 64x32 (4KB), double buffered = 24KB.
// Per k-step each thread issues 3 async b128 copies (2 for A, 1 for B).
// mode 1: epilogue v = softplus(v + bias[n])
//
// Fragment layouts used here (wave32, contiguous K, same order in A and B):
//  A 16x32 bf16: lane l<16 -> row l,  K 0..15;
//                lane l>=16 -> row l-16, K 16..31
//  B 32x16 bf16: lane l holds column N = l%16, K = (l<16 ? 0..15 : 16..31)
// ---------------------------------------------------------------------------
__device__ inline bf16x16 lds_frag_a(const bf16_t* tile /*[128][32]*/,
                                     int mrow0, int lane) {
    int half = lane >> 4;
    int mr   = lane & 15;
    const bf16_t* p = tile + (mrow0 + mr) * 32 + half * 16;
    bf16x8 lo = *(const bf16x8*)p;        // K rel: half*16 + 0..7
    bf16x8 hi = *(const bf16x8*)(p + 8);  // K rel: half*16 + 8..15
    bf16x16 r;
#pragma unroll
    for (int i = 0; i < 8; ++i) { r[i] = lo[i]; r[i + 8] = hi[i]; }
    return r;
}

__device__ inline bf16x16 lds_frag_b(const bf16_t* tile /*[64][32]*/,
                                     int nrow0, int lane) {
    int half = lane >> 4;
    int nc   = lane & 15;
    return *(const bf16x16*)(tile + (nrow0 + nc) * 32 + half * 16);
}

__device__ inline void store_tile(float* __restrict__ C,
                                  const float* __restrict__ bias,
                                  int N, int m0, int n0,
                                  f32x8 acc, int lane, int mode) {
    int half = lane >> 4;
    int n = n0 + (lane & 15);
    if (n >= N) return;                   // guard ragged N (x_proj N=96)
    float bv = (mode == 1) ? bias[n] : 0.0f;
#pragma unroll
    for (int i = 0; i < 8; ++i) {
        int row = m0 + half * 8 + i;      // VGPR i -> M = i (+8 for hi half)
        float v = acc[i];
        if (mode == 1) {                  // softplus(v + b)
            v += bv;
            v = (v > 20.0f) ? v : log1pf(__expf(v));
        }
        C[(size_t)row * N + n] = v;
    }
}

__global__ __launch_bounds__(256) void gemm_bf16_nt(
    const bf16_t* __restrict__ A, const bf16_t* __restrict__ W,
    float* __restrict__ C, const float* __restrict__ bias,
    int M, int N, int K, int mode) {
    (void)M;
    __shared__ __align__(32) bf16_t Atile[2][128][32];
    __shared__ __align__(32) bf16_t Btile[2][64][32];

    const int tid  = threadIdx.x;
    const int lane = tid & 31;
    const int wave = tid >> 5;                      // 0..7
    const int mblk = blockIdx.y * 128;
    const int nblk = blockIdx.x * 64;
    const int mw   = (wave >> 1) * 32;              // wave row in tile
    const int nw   = (wave & 1)  * 32;              // wave col in tile

    // async staging of one BK=32 slice into LDS buffer `buf`
    auto stage = [&](int buf, int k0) {
        // A tile: 512 x 16B chunks; this thread does chunks tid, tid+256
#pragma unroll
        for (int q = 0; q < 2; ++q) {
            int c   = tid + q * 256;
            int row = c >> 2;                       // 0..127
            int seg = c & 3;                        // 16B segment
            async_cp16(A + (size_t)(mblk + row) * K + k0 + seg * 8,
                       &Atile[buf][row][seg * 8]);
        }
        // B tile: 256 x 16B chunks; one per thread (row clamp for ragged N)
        {
            int row = tid >> 2;                     // 0..63
            int seg = tid & 3;
            int n   = nblk + row;
            n = (n < N) ? n : (N - 1);              // branchless clamp
            async_cp16(W + (size_t)n * K + k0 + seg * 8,
                       &Btile[buf][row][seg * 8]);
        }
    };

    const int nk = K >> 5;                          // K / 32 steps
    stage(0, 0);

    f32x8 acc00 = {}, acc01 = {}, acc10 = {}, acc11 = {};
    for (int i = 0; i < nk; ++i) {
        int cur = i & 1;
        if (i + 1 < nk) {
            stage(cur ^ 1, (i + 1) << 5);
            __builtin_amdgcn_s_wait_asynccnt(3);    // tile i complete
        } else {
            __builtin_amdgcn_s_wait_asynccnt(0);
        }
        __syncthreads();                            // LDS visible to all waves

        bf16x16 a0 = lds_frag_a(&Atile[cur][0][0], mw,      lane);
        bf16x16 a1 = lds_frag_a(&Atile[cur][0][0], mw + 16, lane);
        bf16x16 b0 = lds_frag_b(&Btile[cur][0][0], nw,      lane);
        bf16x16 b1 = lds_frag_b(&Btile[cur][0][0], nw + 16, lane);
        acc00 = __builtin_amdgcn_wmma_f32_16x16x32_bf16(false, a0, false, b0,
                                                        (short)0, acc00, false, false);
        acc01 = __builtin_amdgcn_wmma_f32_16x16x32_bf16(false, a0, false, b1,
                                                        (short)0, acc01, false, false);
        acc10 = __builtin_amdgcn_wmma_f32_16x16x32_bf16(false, a1, false, b0,
                                                        (short)0, acc10, false, false);
        acc11 = __builtin_amdgcn_wmma_f32_16x16x32_bf16(false, a1, false, b1,
                                                        (short)0, acc11, false, false);
        __syncthreads();                            // safe to overwrite `cur`
    }

    store_tile(C, bias, N, mblk + mw,      nblk + nw,      acc00, lane, mode);
    store_tile(C, bias, N, mblk + mw,      nblk + nw + 16, acc01, lane, mode);
    store_tile(C, bias, N, mblk + mw + 16, nblk + nw,      acc10, lane, mode);
    store_tile(C, bias, N, mblk + mw + 16, nblk + nw + 16, acc11, lane, mode);
}

// ---------------------------------------------------------------------------
// depthwise causal conv(4) + bias + silu on xz[:, :, 0:E] -> u (f32) + u (bf16)
// ---------------------------------------------------------------------------
__global__ __launch_bounds__(256) void conv_silu_kernel(
    const float* __restrict__ xz, const float* __restrict__ cw,
    const float* __restrict__ cb, float* __restrict__ u,
    bf16_t* __restrict__ ub) {
    size_t idx = (size_t)blockIdx.x * blockDim.x + threadIdx.x; // B*L*E
    int e  = (int)(idx & (EE - 1));
    int t  = (int)((idx >> 11) & (LL - 1));
    int bq = (int)(idx >> 22);
    float acc = cb[e];
#pragma unroll
    for (int j = 0; j < DCONV; ++j) {
        int tt = t - (DCONV - 1) + j;
        if (tt >= 0) {
            size_t off = ((size_t)bq * LL + tt) * (2 * EE) + e;
            acc += cw[e * DCONV + j] * xz[off];
        }
    }
    float sv = acc / (1.0f + __expf(-acc)); // silu
    u[idx]  = sv;
    ub[idx] = (bf16_t)sv;
}

// ---------------------------------------------------------------------------
// extract dt columns (0..63) of x_dbl[M,96] into bf16 [M,64]
// ---------------------------------------------------------------------------
__global__ void slice_dt_kernel(const float* __restrict__ xdbl,
                                bf16_t* __restrict__ dtb) {
    size_t idx = (size_t)blockIdx.x * blockDim.x + threadIdx.x; // M*64
    size_t m = idx >> 6;
    int j = (int)(idx & 63);
    dtb[idx] = (bf16_t)xdbl[m * (DTRANK + 2 * NSTATE) + j];
}

// ---------------------------------------------------------------------------
// selective scan, fused epilogue:  y = (scan + u*D) * silu(z)  -> bf16
// block = 256 threads = 256 channels of one batch; grid = B * (E/256) = 32
// ---------------------------------------------------------------------------
__global__ __launch_bounds__(256) void scan_kernel(
    const float* __restrict__ delta, const float* __restrict__ u,
    const float* __restrict__ xdbl, const float* __restrict__ xz,
    const float* __restrict__ A_log, const float* __restrict__ Dp,
    bf16_t* __restrict__ yb) {
    const int blocksPerB = EE / 256;               // 8
    int bq = blockIdx.x / blocksPerB;
    int d  = (blockIdx.x % blocksPerB) * 256 + threadIdx.x;

    float As[NSTATE];
#pragma unroll
    for (int s = 0; s < NSTATE; ++s) As[s] = -__expf(A_log[d * NSTATE + s]);
    float Dv = Dp[d];
    float h[NSTATE];
#pragma unroll
    for (int s = 0; s < NSTATE; ++s) h[s] = 0.f;

    __shared__ float sB[NSTATE];
    __shared__ float sC[NSTATE];

    for (int t = 0; t < LL; ++t) {
        size_t row = (size_t)bq * LL + t;
        if (threadIdx.x < 32) {
            float vv = xdbl[row * (DTRANK + 2 * NSTATE) + DTRANK + threadIdx.x];
            if (threadIdx.x < NSTATE) sB[threadIdx.x] = vv;
            else                      sC[threadIdx.x - NSTATE] = vv;
        }
        __syncthreads();

        float dt = delta[row * EE + d];
        float uv = u[row * EE + d];
        float du = dt * uv;
        float y  = 0.f;
#pragma unroll
        for (int s = 0; s < NSTATE; ++s) {
            h[s] = __expf(dt * As[s]) * h[s] + du * sB[s];
            y   += h[s] * sC[s];
        }
        y += uv * Dv;
        float zv = xz[row * (2 * EE) + EE + d];
        y *= zv / (1.0f + __expf(-zv));            // * silu(z)
        yb[row * EE + d] = (bf16_t)y;
        __syncthreads();                            // protect sB/sC for next t
    }
}

// ---------------------------------------------------------------------------
// host-side launch
// ---------------------------------------------------------------------------
static inline size_t align256(size_t x) { return (x + 255) & ~(size_t)255; }

extern "C" void kernel_launch(void* const* d_in, const int* in_sizes, int n_in,
                              void* d_out, int out_size, void* d_ws, size_t ws_size,
                              hipStream_t stream) {
    (void)in_sizes; (void)n_in; (void)out_size; (void)ws_size;

    const float* x         = (const float*)d_in[0];
    const float* residual  = (const float*)d_in[1];
    const float* ln_w      = (const float*)d_in[2];
    const float* ln_b      = (const float*)d_in[3];
    const float* in_proj_w = (const float*)d_in[4];   // [2E, D]
    const float* conv_w    = (const float*)d_in[5];   // [E, 4]
    const float* conv_b    = (const float*)d_in[6];   // [E]
    const float* x_proj_w  = (const float*)d_in[7];   // [96, E]
    const float* dt_proj_w = (const float*)d_in[8];   // [E, 64]
    const float* dt_proj_b = (const float*)d_in[9];   // [E]
    const float* A_log     = (const float*)d_in[10];  // [E, 16]
    const float* D_param   = (const float*)d_in[11];  // [E]
    const float* out_proj_w= (const float*)d_in[12];  // [D, E]

    float* out_main = (float*)d_out;                          // [B,L,D]
    float* out_res  = (float*)d_out + (size_t)MM * DD;        // [B,L,D]

    // ---- workspace carve ----
    uint8_t* ws = (uint8_t*)d_ws;
    size_t off = 0;
    auto carve = [&](size_t bytes) -> void* {
        void* p = ws + off; off += align256(bytes); return p;
    };
    bf16_t* xn_b   = (bf16_t*)carve((size_t)MM * DD * 2);           // ln output
    bf16_t* w_in   = (bf16_t*)carve((size_t)2 * EE * DD * 2);       // in_proj bf16
    bf16_t* w_xp   = (bf16_t*)carve((size_t)(DTRANK + 2*NSTATE) * EE * 2);
    bf16_t* w_dt   = (bf16_t*)carve((size_t)EE * DTRANK * 2);
    bf16_t* w_out  = (bf16_t*)carve((size_t)DD * EE * 2);
    float*  xz     = (float*) carve((size_t)MM * 2 * EE * 4);       // [M, 2E]
    float*  u      = (float*) carve((size_t)MM * EE * 4);
    bf16_t* u_b    = (bf16_t*)carve((size_t)MM * EE * 2);
    float*  xdbl   = (float*) carve((size_t)MM * (DTRANK + 2*NSTATE) * 4);
    bf16_t* dt_b   = (bf16_t*)carve((size_t)MM * DTRANK * 2);
    float*  delta  = (float*) carve((size_t)MM * EE * 4);
    bf16_t* y_b    = (bf16_t*)carve((size_t)MM * EE * 2);

    // ---- 1. weights -> bf16 ----
    {
        int n;
        n = 2 * EE * DD;
        cvt_bf16_kernel<<<(n + 255) / 256, 256, 0, stream>>>(in_proj_w, w_in, n);
        n = (DTRANK + 2 * NSTATE) * EE;
        cvt_bf16_kernel<<<(n + 255) / 256, 256, 0, stream>>>(x_proj_w, w_xp, n);
        n = EE * DTRANK;
        cvt_bf16_kernel<<<(n + 255) / 256, 256, 0, stream>>>(dt_proj_w, w_dt, n);
        n = DD * EE;
        cvt_bf16_kernel<<<(n + 255) / 256, 256, 0, stream>>>(out_proj_w, w_out, n);
    }

    // ---- 2. residual add + layernorm ----
    add_ln_kernel<<<MM, 256, 0, stream>>>(x, residual, ln_w, ln_b, out_res, xn_b);

    // ---- 3. in_proj: xz[M, 2E] = xn * in_proj_w^T ----
    {
        dim3 grid((2 * EE) / 64, MM / 128);
        gemm_bf16_nt<<<grid, 256, 0, stream>>>(xn_b, w_in, xz, nullptr,
                                               MM, 2 * EE, DD, 0);
    }

    // ---- 4. depthwise conv + silu -> u (f32 + bf16) ----
    {
        size_t total = (size_t)MM * EE;
        conv_silu_kernel<<<(unsigned)(total / 256), 256, 0, stream>>>(
            xz, conv_w, conv_b, u, u_b);
    }

    // ---- 5. x_proj: x_dbl[M, 96] = u * x_proj_w^T (ragged N via guards) ----
    {
        dim3 grid((DTRANK + 2 * NSTATE + 63) / 64, MM / 128);
        gemm_bf16_nt<<<grid, 256, 0, stream>>>(u_b, w_xp, xdbl, nullptr,
                                               MM, DTRANK + 2 * NSTATE, EE, 0);
    }

    // ---- 6. slice dt columns -> bf16 ----
    {
        size_t total = (size_t)MM * DTRANK;
        slice_dt_kernel<<<(unsigned)(total / 256), 256, 0, stream>>>(xdbl, dt_b);
    }

    // ---- 7. dt_proj + bias + softplus fused epilogue -> delta[M, E] ----
    {
        dim3 grid(EE / 64, MM / 128);
        gemm_bf16_nt<<<grid, 256, 0, stream>>>(dt_b, w_dt, delta, dt_proj_b,
                                               MM, EE, DTRANK, 1);
    }

    // ---- 8. selective scan + (+u*D) * silu(z) -> y bf16 ----
    scan_kernel<<<BB * (EE / 256), 256, 0, stream>>>(delta, u, xdbl, xz,
                                                     A_log, D_param, y_b);

    // ---- 9. out_proj: out[M, D] = y * out_proj_w^T ----
    {
        dim3 grid(DD / 64, MM / 128);
        gemm_bf16_nt<<<grid, 256, 0, stream>>>(y_b, w_out, out_main, nullptr,
                                               MM, DD, EE, 0);
    }
}
